// DynamicFcGagaLayer_72679436582925
// MI455X (gfx1250) — matrix operand order repaired and checked by reading, not compiled, not run on hardware
//
#include <hip/hip_runtime.h>
#include <hip/hip_bf16.h>

// ---------------- problem constants ----------------
#define BB    8
#define NN    512
#define TT    24
#define HID   256
#define HOR   24
#define NID   64
#define NBLK  3
#define DD    (TT + NN*TT + NID)   // 12376
#define ROWS  (BB*NN)              // 4096

// ---------------- WMMA types ----------------
typedef __attribute__((ext_vector_type(16))) __bf16 bf16x16;
typedef __attribute__((ext_vector_type(8)))  float  f32x8;
typedef __attribute__((ext_vector_type(2)))  float  f32x2;
typedef __attribute__((ext_vector_type(2)))  __bf16 bf16x2;

union FragU { bf16x16 v; unsigned u[8]; };

// GEMM tiling: block tile 128x64, 8 waves, each wave owns a 32x32 output tile
#define BM 128
#define BN 64
#define BK 32
#define KP (BK/2)        // 16 K-pairs per step
#define LDA_SU (KP + 1)  // A LDS row stride in dwords (pad)
#define LDB_SU (BN + 1)  // B LDS row stride in dwords (pad)

// one v_cvt_pk_bf16_f32: float2 -> packed bf16x2 dword
__device__ __forceinline__ unsigned pk_bf16(float a, float b) {
    f32x2 s = {a, b};
    bf16x2 t = __builtin_convertvector(s, bf16x2);
    unsigned r;
    __builtin_memcpy(&r, &t, 4);
    return r;   // low 16 = a (even K), high 16 = b (odd K)
}

// A fragment (16x32): lane = half*16 + l, row M = l.
// dword i in 0..7 : K-pair = half*8 + i
__device__ __forceinline__ bf16x16 load_a_frag(const unsigned* AsU, int row, int half) {
    FragU f;
    const unsigned* p = AsU + row * LDA_SU + half * 8;
#pragma unroll
    for (int i = 0; i < 4; ++i) {
        f.u[i]     = p[i];
        f.u[4 + i] = p[4 + i];
    }
    return f.v;
}

// B fragment (32x16): col N = l, dword i: K-pair = half*8 + i
__device__ __forceinline__ bf16x16 load_b_frag(const unsigned* BsU, int col, int half) {
    FragU f;
    const unsigned* p = BsU + half * 8 * LDB_SU + col;
#pragma unroll
    for (int i = 0; i < 8; ++i) f.u[i] = p[i * LDB_SU];
    return f.v;
}

__device__ __forceinline__ void mma_step(const unsigned* AsU, const unsigned* BsU,
                                         int mW, int nW, int l, int half,
                                         f32x8& a00, f32x8& a01, f32x8& a10, f32x8& a11) {
    bf16x16 fa0 = load_a_frag(AsU, mW + l, half);
    bf16x16 fb0 = load_b_frag(BsU, nW + l, half);
    bf16x16 fb1 = load_b_frag(BsU, nW + 16 + l, half);
    a00 = __builtin_amdgcn_wmma_f32_16x16x32_bf16(false, fa0, false, fb0, (short)0, a00, false, false);
    a01 = __builtin_amdgcn_wmma_f32_16x16x32_bf16(false, fa0, false, fb1, (short)0, a01, false, false);
    bf16x16 fa1 = load_a_frag(AsU, mW + 16 + l, half);
    a10 = __builtin_amdgcn_wmma_f32_16x16x32_bf16(false, fa1, false, fb0, (short)0, a10, false, false);
    a11 = __builtin_amdgcn_wmma_f32_16x16x32_bf16(false, fa1, false, fb1, (short)0, a11, false, false);
}

// epilogue for one 16x16 fragment; acc BY VALUE (never take accumulator addresses)
template <int MODE>
__device__ __forceinline__ void epilogue_frag(float* __restrict__ C,
                                              const float* __restrict__ bias,
                                              int ldc, int Ndim,
                                              int row0, int gc, f32x8 acc) {
    if (gc < Ndim) {
        const float bv = bias[gc];
#pragma unroll
        for (int i = 0; i < 8; ++i) {
            const size_t o = (size_t)(row0 + i) * ldc + gc;
            if (MODE == 0) C[o] = fmaxf(acc[i] + bv, 0.0f);
            else           C[o] = fmaxf(C[o] - (acc[i] + bv), 0.0f);
        }
    }
}

// Generic GEMM: acc = A(MxK,f32,lda) * Bw(KxN,f32,ldb) ; BF16 inputs, F32 accum.
// MODE 0: C = relu(acc + bias)          MODE 1: C = relu(C - (acc + bias)) in place
template <int MODE>
__global__ __launch_bounds__(256, 1) void gemm_wmma_kernel(
    const float* __restrict__ A, int lda,
    const float* __restrict__ Bw, int ldb,
    const float* __restrict__ bias,
    float* __restrict__ C, int ldc,
    int K, int Ndim)
{
    __shared__ unsigned AsU[2][BM * LDA_SU];   // ping-pong
    __shared__ unsigned BsU[2][KP * LDB_SU];

    const int tid  = threadIdx.x;
    const int wave = tid >> 5;
    const int lane = tid & 31;
    const int half = lane >> 4;
    const int l    = lane & 15;

    const int m0 = blockIdx.x * BM;
    const int n0 = blockIdx.y * BN;
    const int mW = (wave & 3) * 32;   // 4 waves over M
    const int nW = (wave >> 2) * 32;  // 2 waves over N

    // loop-invariant staging geometry (single base + per-slot stride)
    const float* aBase = A + (size_t)(m0 + (tid >> 3)) * lda + (tid & 7) * 4;
    const size_t aStep = (size_t)32 * lda;                 // 32 rows per slot
    const int    aSb   = (tid >> 3) * LDA_SU + (tid & 7) * 2;

    const int  kp   = tid >> 4;             // K-pair row 0..15
    const int  cq2  = tid & 15;             // col quad
    const int  gn   = n0 + cq2 * 4;
    const bool gnOK = gn < Ndim;            // Ndim % 4 == 0: quad never straddles
    const float* bP0 = Bw + (size_t)(kp * 2) * ldb + gn;
    const float* bP1 = bP0 + ldb;
    const int  bSo  = kp * LDB_SU + cq2 * 4;

    unsigned* Acur  = AsU[0]; unsigned* Anext = AsU[1];
    unsigned* Bcur  = BsU[0]; unsigned* Bnext = BsU[1];

    f32x8 a00 = {0.f,0.f,0.f,0.f,0.f,0.f,0.f,0.f};
    f32x8 a01 = a00, a10 = a00, a11 = a00;

    float4 a0r, a1r, a2r, a3r, b0r, b1r;

#define LDTILE(kk_)                                                        \
    do { const int kk__ = (kk_);                                           \
        a0r = *(const float4*)(aBase + kk__);                              \
        a1r = *(const float4*)(aBase + aStep + kk__);                      \
        a2r = *(const float4*)(aBase + 2 * aStep + kk__);                  \
        a3r = *(const float4*)(aBase + 3 * aStep + kk__);                  \
        if (gnOK) {                                                        \
            b0r = *(const float4*)(bP0 + (size_t)kk__ * ldb);              \
            b1r = *(const float4*)(bP1 + (size_t)kk__ * ldb);              \
        } else {                                                           \
            b0r = make_float4(0.f,0.f,0.f,0.f); b1r = b0r;                 \
        }                                                                  \
    } while (0)

#define STTILE(dst_a, dst_b)                                               \
    do {                                                                   \
        (dst_a)[aSb]                   = pk_bf16(a0r.x, a0r.y);            \
        (dst_a)[aSb + 1]               = pk_bf16(a0r.z, a0r.w);            \
        (dst_a)[aSb + 32*LDA_SU]       = pk_bf16(a1r.x, a1r.y);            \
        (dst_a)[aSb + 32*LDA_SU + 1]   = pk_bf16(a1r.z, a1r.w);            \
        (dst_a)[aSb + 64*LDA_SU]       = pk_bf16(a2r.x, a2r.y);            \
        (dst_a)[aSb + 64*LDA_SU + 1]   = pk_bf16(a2r.z, a2r.w);            \
        (dst_a)[aSb + 96*LDA_SU]       = pk_bf16(a3r.x, a3r.y);            \
        (dst_a)[aSb + 96*LDA_SU + 1]   = pk_bf16(a3r.z, a3r.w);            \
        unsigned* bp__ = (dst_b) + bSo;                                    \
        bp__[0] = pk_bf16(b0r.x, b1r.x);                                   \
        bp__[1] = pk_bf16(b0r.y, b1r.y);                                   \
        bp__[2] = pk_bf16(b0r.z, b1r.z);                                   \
        bp__[3] = pk_bf16(b0r.w, b1r.w);                                   \
    } while (0)

    const int kfull = K / BK;
    if (kfull > 0) {
        LDTILE(0);
        STTILE(Acur, Bcur);
        __syncthreads();
        for (int ks = 0; ks < kfull; ++ks) {
            const bool more = (ks + 1 < kfull);
            if (more) LDTILE((ks + 1) * BK);                 // loads in flight during WMMA
            if (lane == 0 && ks + 2 < kfull)
                __builtin_prefetch(aBase + (size_t)(ks + 2) * BK, 0, 1);
            mma_step(Acur, Bcur, mW, nW, l, half, a00, a01, a10, a11);
            if (more) STTILE(Anext, Bnext);
            __syncthreads();                                  // single barrier per step
            unsigned* t_;
            t_ = Acur; Acur = Anext; Anext = t_;
            t_ = Bcur; Bcur = Bnext; Bnext = t_;
        }
    }

    // K tail (only GEMM1: 12376 = 386*32 + 24), guarded scalar staging
    if (K & (BK - 1)) {
        const int kk = kfull * BK;
#pragma unroll
        for (int q = 0; q < 8; ++q) {                        // 128 rows x 16 pairs
            const int idx = tid + q * 256;
            const int rA  = idx >> 4;
            const int kpi = idx & 15;
            const int gk  = kk + kpi * 2;
            const float* p = A + (size_t)(m0 + rA) * lda + gk;
            float v0 = (gk     < K) ? p[0] : 0.0f;
            float v1 = (gk + 1 < K) ? p[1] : 0.0f;
            Acur[rA * LDA_SU + kpi] = pk_bf16(v0, v1);
        }
#pragma unroll
        for (int q = 0; q < 4; ++q) {                        // 16 pairs x 64 cols
            const int idx = tid + q * 256;
            const int kpi = idx >> 6;
            const int cB  = idx & 63;
            const int gk  = kk + kpi * 2;
            const int gc  = n0 + cB;
            float v0 = (gk     < K && gc < Ndim) ? Bw[(size_t)gk * ldb + gc]       : 0.0f;
            float v1 = (gk + 1 < K && gc < Ndim) ? Bw[(size_t)(gk + 1) * ldb + gc] : 0.0f;
            Bcur[kpi * LDB_SU + cB] = pk_bf16(v0, v1);
        }
        __syncthreads();
        mma_step(Acur, Bcur, mW, nW, l, half, a00, a01, a10, a11);
    }

    // epilogue: 4 fragments, accumulators passed by value
    epilogue_frag<MODE>(C, bias, ldc, Ndim, m0 + mW +      8 * half, n0 + nW +      l, a00);
    epilogue_frag<MODE>(C, bias, ldc, Ndim, m0 + mW +      8 * half, n0 + nW + 16 + l, a01);
    epilogue_frag<MODE>(C, bias, ldc, Ndim, m0 + mW + 16 + 8 * half, n0 + nW +      l, a10);
    epilogue_frag<MODE>(C, bias, ldc, Ndim, m0 + mW + 16 + 8 * half, n0 + nW + 16 + l, a11);
#undef LDTILE
#undef STTILE
}

// level = max_t hist, recip = div_no_nan ; also zero the forecast accumulator
__global__ void levels_kernel(const float* __restrict__ hist,
                              float* __restrict__ Lv, float* __restrict__ Rc,
                              float* __restrict__ F)
{
    int r = blockIdx.x * blockDim.x + threadIdx.x;
    if (r < ROWS) {
        const float* h = hist + (size_t)r * TT;
        float mx = h[0];
#pragma unroll
        for (int t = 1; t < TT; ++t) mx = fmaxf(mx, h[t]);
        Lv[r] = mx;
        Rc[r] = (mx == 0.0f) ? 0.0f : 1.0f / mx;
#pragma unroll
        for (int o = 0; o < HOR; ++o) F[(size_t)r * HOR + o] = 0.0f;
    }
}

// Build X[r, 0:D] = [hist/level | relu((adj*hist_m - level)/level) | node_emb]
__global__ __launch_bounds__(256) void prep_kernel(
    const float* __restrict__ hist, const int* __restrict__ nid,
    const float* __restrict__ adj,  const float* __restrict__ emb,
    const float* __restrict__ Lv,   const float* __restrict__ Rc,
    float* __restrict__ X)
{
    const int r = blockIdx.x;           // 0..4095
    const int b = r >> 9;
    const int n = r & (NN - 1);
    const float level = Lv[r];
    const float rcp   = Rc[r];
    const float* histB  = hist + (size_t)b * NN * TT;
    const float* adjRow = adj + ((size_t)b * NN + n) * NN;
    const int    id     = nid[r];
    float* Xrow = X + (size_t)r * DD;

    for (int c = threadIdx.x; c < DD; c += blockDim.x) {
        float v;
        if (c < TT) {
            v = histB[n * TT + c] * rcp;
        } else if (c < TT + NN * TT) {
            const int idx = c - TT;
            const int m = idx / TT;
            const int t = idx - m * TT;
            const float a = adjRow[m] * histB[m * TT + t];
            v = fmaxf((a - level) * rcp, 0.0f);   // rcp==0 when level==0 -> div_no_nan
        } else {
            v = emb[id * NID + (c - (TT + NN * TT))];
        }
        Xrow[c] = v;
    }
}

// F += H(4096x256) @ Wf(256x24) + bf   (tiny: 50 MFLOP, VALU is fine)
__global__ void forecast_kernel(const float* __restrict__ H,
                                const float* __restrict__ Wf,
                                const float* __restrict__ bfv,
                                float* __restrict__ F)
{
    int g = blockIdx.x * blockDim.x + threadIdx.x;
    if (g < ROWS * HOR) {
        const int r = g / HOR;
        const int o = g - r * HOR;
        const float* h = H + (size_t)r * HID;
        float acc = bfv[o];
#pragma unroll 8
        for (int k = 0; k < HID; ++k) acc = fmaf(h[k], Wf[k * HOR + o], acc);
        F[g] += acc;
    }
}

__global__ void finalize_kernel(const float* __restrict__ F,
                                const float* __restrict__ Lv,
                                float* __restrict__ out)
{
    int g = blockIdx.x * blockDim.x + threadIdx.x;
    if (g < ROWS * HOR) {
        const int r = g / HOR;
        out[g] = F[g] * Lv[r];
    }
}

extern "C" void kernel_launch(void* const* d_in, const int* in_sizes, int n_in,
                              void* d_out, int out_size, void* d_ws, size_t ws_size,
                              hipStream_t stream) {
    (void)in_sizes; (void)n_in; (void)out_size; (void)ws_size;
    const float* hist  = (const float*)d_in[0];
    const int*   nid   = (const int*)  d_in[1];
    /* d_in[2] time_of_day_in unused by reference */
    const float* adj   = (const float*)d_in[3];
    const float* emb   = (const float*)d_in[4];
    const float* W_in  = (const float*)d_in[5];
    const float* b_in  = (const float*)d_in[6];
    const float* W_hid = (const float*)d_in[7];
    const float* b_hid = (const float*)d_in[8];
    const float* W_f   = (const float*)d_in[9];
    const float* b_f   = (const float*)d_in[10];
    const float* W_b   = (const float*)d_in[11];
    const float* b_b   = (const float*)d_in[12];

    // d_out = [backcast (ROWS x D) | forecast (ROWS x HOR)]
    // The backcast region doubles as the live X buffer across blocks.
    float* X    = (float*)d_out;
    float* outF = X + (size_t)ROWS * DD;

    // workspace layout (~8.8 MB)
    char* w = (char*)d_ws;
    float* Lv = (float*)(w);                         // 16 KB slot
    float* Rc = (float*)(w + 16384);                 // 16 KB slot
    float* H1 = (float*)(w + 32768);                 // 4 MB
    float* H2 = (float*)(w + 32768 + (size_t)ROWS*HID*4);
    float* F  = (float*)(w + 32768 + (size_t)2*ROWS*HID*4);

    levels_kernel<<<ROWS/256, 256, 0, stream>>>(hist, Lv, Rc, F);
    prep_kernel<<<ROWS, 256, 0, stream>>>(hist, nid, adj, emb, Lv, Rc, X);

    const dim3 gridSq(ROWS/BM, HID/BN);                 // (32, 4)
    const dim3 gridBc(ROWS/BM, (DD + BN - 1)/BN);       // (32, 194)

    for (int i = 0; i < NBLK; ++i) {
        const float* Wi  = W_in  + (size_t)i * DD * HID;
        const float* bi  = b_in  + (size_t)i * HID;
        const float* Wh0 = W_hid + ((size_t)i * 2 + 0) * HID * HID;
        const float* bh0 = b_hid + ((size_t)i * 2 + 0) * HID;
        const float* Wh1 = W_hid + ((size_t)i * 2 + 1) * HID * HID;
        const float* bh1 = b_hid + ((size_t)i * 2 + 1) * HID;
        const float* Wfi = W_f   + (size_t)i * HID * HOR;
        const float* bfi = b_f   + (size_t)i * HOR;
        const float* Wbi = W_b   + (size_t)i * HID * DD;
        const float* bbi = b_b   + (size_t)i * DD;

        // h = relu(X @ W_in + b_in)          M=4096 K=12376 N=256
        gemm_wmma_kernel<0><<<gridSq, 256, 0, stream>>>(X, DD, Wi, HID, bi, H1, HID, DD, HID);
        // two hidden layers 256x256
        gemm_wmma_kernel<0><<<gridSq, 256, 0, stream>>>(H1, HID, Wh0, HID, bh0, H2, HID, HID, HID);
        gemm_wmma_kernel<0><<<gridSq, 256, 0, stream>>>(H2, HID, Wh1, HID, bh1, H1, HID, HID, HID);
        // forecast accumulate (tiny)
        forecast_kernel<<<(ROWS*HOR + 255)/256, 256, 0, stream>>>(H1, Wfi, bfi, F);
        // backcast: X = relu(X - (h @ W_b + b_b))   M=4096 K=256 N=12376
        gemm_wmma_kernel<1><<<gridBc, 256, 0, stream>>>(H1, HID, Wbi, DD, bbi, X, DD, HID, DD);
    }

    finalize_kernel<<<(ROWS*HOR + 255)/256, 256, 0, stream>>>(F, Lv, outF);
}
